// TermEmbedder_15779709845506
// MI455X (gfx1250) — matrix operand order repaired and checked by reading, not compiled, not run on hardware
//
#include <hip/hip_runtime.h>

typedef __attribute__((ext_vector_type(16))) __bf16 bf16x16;
typedef __attribute__((ext_vector_type(8)))  float  f32x8;

#define B_   128
#define H_   256
#define NN_  1023
#define AST  40   // LDS row stride (bf16 elems) for A tile: 80 bytes
#define BST  40   // LDS row stride for B tile
#define SA_ELT (64 * AST)     // one A buffer, bf16 elems
#define SB_ELT (320 * BST)    // one B buffer, bf16 elems

union Frag { bf16x16 v; uint4 q[2]; };

__device__ __forceinline__ unsigned short f2bf(float f) {
  unsigned int u = __float_as_uint(f);
  u += 0x7FFFu + ((u >> 16) & 1u);          // round to nearest even
  return (unsigned short)(u >> 16);
}
__device__ __forceinline__ float bf2f(unsigned short s) {
  return __uint_as_float(((unsigned int)s) << 16);
}
__device__ __forceinline__ float sigmoidf(float x) {
  return 1.0f / (1.0f + __expf(-x));
}
__device__ __forceinline__ float tanh_f(float x) {
  float e = __expf(2.0f * x);
  return 1.0f - 2.0f / (e + 1.0f);
}

// gfx1250 async global->LDS copy (16B per lane), tracked by ASYNCcnt.
__device__ __forceinline__ void async_g2l_b128(unsigned lds_off, const void* gaddr) {
  asm volatile("global_load_async_to_lds_b128 %0, %1, off"
               :: "v"(lds_off), "v"(gaddr) : "memory");
}

// ---------------------------------------------------------------------------
// Weight prep: Wx/Ul/Ur (H x 5H, f32 row-major) -> bf16 transposed [5H][H]
// ---------------------------------------------------------------------------
__global__ __launch_bounds__(256) void prep_weights(
    const float* __restrict__ Wx, const float* __restrict__ Ul,
    const float* __restrict__ Ur, unsigned short* __restrict__ WxT,
    unsigned short* __restrict__ UlT, unsigned short* __restrict__ UrT) {
  int idx = blockIdx.x * 256 + threadIdx.x;
  const int per = 5 * H_ * H_;               // 1280*256
  if (idx >= 3 * per) return;
  int w = idx / per, rem = idx % per;
  int col = rem / H_, k = rem % H_;
  const float* S = (w == 0) ? Wx : ((w == 1) ? Ul : Ur);
  unsigned short* D = (w == 0) ? WxT : ((w == 1) ? UlT : UrT);
  D[col * H_ + k] = f2bf(S[k * (5 * H_) + col]);
}

// ---------------------------------------------------------------------------
// Per-level token-embedding gather: x_bf[(node*B + b), :] = bf16(emb[token])
// One thread per 8 elements (two float4 loads -> one uint4 store).
// ---------------------------------------------------------------------------
__global__ __launch_bounds__(256) void gather_x(
    const int* __restrict__ tokens, const float* __restrict__ emb,
    unsigned short* __restrict__ x_bf, int start, int n) {
  int idx = blockIdx.x * 256 + threadIdx.x;
  int total = n * B_ * (H_ / 8);
  if (idx >= total) return;
  int row = idx >> 5;            // H/8 = 32 chunks per row
  int chunk = idx & 31;
  int b = row & 127, node = row >> 7;
  int token = tokens[b * NN_ + start + node];
  const float4* e = (const float4*)(emb + (size_t)token * H_ + chunk * 8);
  float4 f0 = e[0], f1 = e[1];
  uint4 o;
  o.x = (unsigned)f2bf(f0.x) | ((unsigned)f2bf(f0.y) << 16);
  o.y = (unsigned)f2bf(f0.z) | ((unsigned)f2bf(f0.w) << 16);
  o.z = (unsigned)f2bf(f1.x) | ((unsigned)f2bf(f1.y) << 16);
  o.w = (unsigned)f2bf(f1.z) | ((unsigned)f2bf(f1.w) << 16);
  *(uint4*)(x_bf + (size_t)row * H_ + chunk * 8) = o;
}

// ---------------------------------------------------------------------------
// One tree level: gates = X*Wx (+ HL*Ul + HR*Ur) + b, then LSTM cell.
// Activations node-major: row m = node*B + b (child rows of one parent tile
// are contiguous since B=128 >= 64-row tile).
// Block: 256 threads = 8 waves (2 M-groups x 4 N-groups).
// Tile: 64 rows x 64 hidden-cols x 5 gates; 10 WMMAs / wave / K-step.
// Double-buffered LDS; all staging via async global->LDS (ASYNCcnt).
// ---------------------------------------------------------------------------
__global__ __launch_bounds__(256) void tree_lstm_level(
    const unsigned short* __restrict__ x_bf,    // (n*B, H) bf16, node-major
    const unsigned short* __restrict__ WxT,     // (5H, H) bf16, K-contig
    const unsigned short* __restrict__ UlT,
    const unsigned short* __restrict__ UrT,
    const float* __restrict__ bias,             // (5H)
    const unsigned short* __restrict__ h_child, // (2n*B, H) bf16
    const float* __restrict__ c_child,          // (2n*B, H) f32
    unsigned short* __restrict__ h_out,         // (n*B, H) bf16
    float* __restrict__ c_out,                  // (n*B, H) f32
    int hasChild) {
  __shared__ __align__(16) unsigned short sA[2 * SA_ELT];   // 2 x  5120 B
  __shared__ __align__(16) unsigned short sB[2 * SB_ELT];   // 2 x 25600 B

  const int tid  = threadIdx.x;
  const int lane = tid & 31;
  const int wave = tid >> 5;
  const int half = lane >> 4;
  const int lr   = lane & 15;
  const int wm   = wave & 1;   // 0..1: 32-row M-group
  const int wn   = wave >> 1;  // 0..3: 16-col N-group

  const int tile_m = blockIdx.x * 64;
  const int n_base = blockIdx.y * 64;
  const int p  = tile_m >> 7;       // node index
  const int b0 = tile_m & 127;

  const unsigned sA_off = (unsigned)(uintptr_t)(void*)&sA[0];
  const unsigned sB_off = (unsigned)(uintptr_t)(void*)&sB[0];

  f32x8 acc[2][5] = {};

  const int nsteps = (hasChild ? 3 : 1) * 8;   // 32-K steps

  // Stage step's 64x32 A tile + 320x32 B tile into LDS buffer `buf`.
  // Per wave: 5 async (B) + 1 async (A) = 6 ASYNCcnt increments.
  auto stage = [&](int step, int buf) {
    int s  = step >> 3;
    int kk = (step & 7) << 5;
    const unsigned short* WT = (s == 0) ? WxT : ((s == 1) ? UlT : UrT);
    const unsigned short* Abase =
        (s == 0) ? (x_bf + (size_t)tile_m * H_)
                 : (h_child + (size_t)((2 * p + (s - 1)) * B_ + b0) * H_);
    unsigned sBo = sB_off + (unsigned)buf * (SB_ELT * 2);
    unsigned sAo = sA_off + (unsigned)buf * (SA_ELT * 2);
    for (int ch = tid; ch < 1280; ch += 256) {
      int row = ch >> 2, part = ch & 3;
      int g = row >> 6, cj = row & 63;
      int gw = g * H_ + n_base + cj;
      async_g2l_b128(sBo + (unsigned)(row * BST + part * 8) * 2,
                     WT + (size_t)gw * H_ + kk + part * 8);
    }
    {
      int row = tid >> 2, part = tid & 3;
      async_g2l_b128(sAo + (unsigned)(row * AST + part * 8) * 2,
                     Abase + (size_t)row * H_ + kk + part * 8);
    }
  };

  stage(0, 0);
  for (int step = 0; step < nsteps; ++step) {
    const int cur = step & 1;
    if (step + 1 < nsteps) {
      stage(step + 1, cur ^ 1);                 // prefetch under compute
      asm volatile("s_wait_asynccnt 0x6" ::: "memory");  // cur buffer done
    } else {
      asm volatile("s_wait_asynccnt 0x0" ::: "memory");
    }
    __syncthreads();

    const unsigned short* cA = &sA[cur * SA_ELT];
    const unsigned short* cB = &sB[cur * SB_ELT];

    // two A fragments (16x32 bf16, gfx1250 layout)
    Frag a0, a1;
    {
      int ar0 = wm * 32 + lr;
      int ar1 = ar0 + 16;
      a0.q[0] = *(const uint4*)(&cA[ar0 * AST + half * 8]);
      a0.q[1] = *(const uint4*)(&cA[ar0 * AST + 16 + half * 8]);
      a1.q[0] = *(const uint4*)(&cA[ar1 * AST + half * 8]);
      a1.q[1] = *(const uint4*)(&cA[ar1 * AST + 16 + half * 8]);
    }
    // 5 B fragments, each reused by both A fragments
#pragma unroll
    for (int g = 0; g < 5; ++g) {
      Frag bf;
      int brow = g * 64 + wn * 16 + lr;                        // N = lane col
      const uint4* bp = (const uint4*)(&cB[brow * BST + half * 8]);
      bf.q[0] = bp[0];                                         // K = half*8 ..+7
      bf.q[1] = bp[2];                                         // K = 16+half*8..+7
      acc[0][g] = __builtin_amdgcn_wmma_f32_16x16x32_bf16(
          false, a0.v, false, bf.v, (short)0, acc[0][g], false, false);
      acc[1][g] = __builtin_amdgcn_wmma_f32_16x16x32_bf16(
          false, a1.v, false, bf.v, (short)0, acc[1][g], false, false);
    }
    __syncthreads();   // all waves done reading `cur` before it is re-staged
  }

  // ---- epilogue: bias + LSTM cell ----
  const int j = n_base + wn * 16 + lr;       // hidden column 0..255
  const float b_i  = bias[0 * H_ + j];
  const float b_fl = bias[1 * H_ + j];
  const float b_fr = bias[2 * H_ + j];
  const float b_o  = bias[3 * H_ + j];
  const float b_u  = bias[4 * H_ + j];
  const int clbase = (2 * p) * B_;

#pragma unroll
  for (int t = 0; t < 2; ++t) {
    const int mrow0 = tile_m + wm * 32 + t * 16 + half * 8;
#pragma unroll
    for (int r = 0; r < 8; ++r) {
      int m = mrow0 + r;
      float gi = sigmoidf(acc[t][0][r] + b_i);
      float gl = sigmoidf(acc[t][1][r] + b_fl);
      float gr = sigmoidf(acc[t][2][r] + b_fr);
      float go = sigmoidf(acc[t][3][r] + b_o);
      float gu = tanh_f (acc[t][4][r] + b_u);
      float cn = gi * gu;
      if (hasChild) {
        int b = m & 127;
        int clrow = clbase + b;                       // left child row
        float cl = c_child[(size_t)clrow * H_ + j];
        float cr = c_child[(size_t)(clrow + B_) * H_ + j];
        cn += gl * cl + gr * cr;
      }
      float hh = go * tanh_f(cn);
      c_out[(size_t)m * H_ + j] = cn;
      h_out[(size_t)m * H_ + j] = f2bf(hh);
    }
  }
}

// ---------------------------------------------------------------------------
// Root h (bf16, rows 0..B-1) -> f32 output (B, H)
// ---------------------------------------------------------------------------
__global__ __launch_bounds__(256) void write_out(
    const unsigned short* __restrict__ hroot, float* __restrict__ out) {
  int i = blockIdx.x * 256 + threadIdx.x;
  if (i < B_ * H_) out[i] = bf2f(hroot[i]);
}

// ---------------------------------------------------------------------------
extern "C" void kernel_launch(void* const* d_in, const int* in_sizes, int n_in,
                              void* d_out, int out_size, void* d_ws, size_t ws_size,
                              hipStream_t stream) {
  (void)in_sizes; (void)n_in; (void)out_size; (void)ws_size;
  const int*   tokens = (const int*)  d_in[0];
  const float* emb    = (const float*)d_in[1];
  const float* Wx     = (const float*)d_in[2];
  const float* Ul     = (const float*)d_in[3];
  const float* Ur     = (const float*)d_in[4];
  const float* bias   = (const float*)d_in[5];

  char* ws = (char*)d_ws;
  size_t off = 0;
  auto take = [&](size_t bytes) -> char* {
    char* pp = ws + off;
    off += (bytes + 255) & ~(size_t)255;
    return pp;
  };
  unsigned short* WxT = (unsigned short*)take(5 * H_ * H_ * 2);
  unsigned short* UlT = (unsigned short*)take(5 * H_ * H_ * 2);
  unsigned short* UrT = (unsigned short*)take(5 * H_ * H_ * 2);
  unsigned short* Xb  = (unsigned short*)take((size_t)65536 * H_ * 2); // max level
  // ping-pong level buffers: buf0 = even levels (max 256 nodes = 32768 rows),
  //                          buf1 = odd  levels (max 512 nodes = 65536 rows)
  unsigned short* Hb[2];
  float*          Cb[2];
  Hb[0] = (unsigned short*)take((size_t)32768 * H_ * 2);
  Cb[0] = (float*)         take((size_t)32768 * H_ * 4);
  Hb[1] = (unsigned short*)take((size_t)65536 * H_ * 2);
  Cb[1] = (float*)         take((size_t)65536 * H_ * 4);

  {
    int total = 3 * 5 * H_ * H_;
    prep_weights<<<(total + 255) / 256, 256, 0, stream>>>(Wx, Ul, Ur, WxT, UlT, UrT);
  }

  for (int d = 9; d >= 0; --d) {
    int n = 1 << d;
    int start = n - 1;
    int hasChild = (d != 9) ? 1 : 0;
    int outb = d & 1;                        // level-9 -> buf1 (big), alternate down

    int gthreads = n * B_ * (H_ / 8);
    gather_x<<<(gthreads + 255) / 256, 256, 0, stream>>>(tokens, emb, Xb, start, n);

    dim3 grid(n * 2, 4);                     // M/64 x H/64 tiles
    tree_lstm_level<<<grid, 256, 0, stream>>>(
        Xb, WxT, UlT, UrT, bias,
        Hb[outb ^ 1], Cb[outb ^ 1],          // children (unused at leaf level)
        Hb[outb], Cb[outb], hasChild);
  }

  write_out<<<(B_ * H_ + 255) / 256, 256, 0, stream>>>(Hb[0], (float*)d_out);
}
